// LiquidCell_15195594293730
// MI455X (gfx1250) — matrix operand order repaired and checked by reading, not compiled, not run on hardware
//
#include <hip/hip_runtime.h>
#include <hip/hip_bf16.h>
#include <math.h>

typedef __attribute__((ext_vector_type(16))) __bf16 v16bf;
typedef __attribute__((ext_vector_type(8)))  float  v8f;

#define D_IN   128
#define D_H    128
#define KTOT   256          // concat(x, state)
#define NBLK   131072       // B
#define LN_EPSF 1e-5f

// ---- fast transcendentals: gfx1250 has V_TANH_F32 (TRANS pipe) ----
#if defined(__has_builtin)
#if __has_builtin(__builtin_amdgcn_tanhf)
#define FAST_TANH(x) __builtin_amdgcn_tanhf(x)
#endif
#if __has_builtin(__builtin_amdgcn_rcpf)
#define FAST_RCP(x) __builtin_amdgcn_rcpf(x)
#endif
#endif
#ifndef FAST_TANH
#define FAST_TANH(x) tanhf(x)
#endif
#ifndef FAST_RCP
#define FAST_RCP(x) (1.0f / (x))
#endif
#define FAST_SIGMOID(x) FAST_RCP(1.0f + __expf(-(x)))

// LDS fragment storage (halfword = __bf16 units)
// 8 k-chunks * 8 n-tiles * 32 lanes * 16 halfwords = 32768 halfwords per weight matrix
#define WD_OFF 0
#define WG_OFF (8 * 8 * 32 * 16)
#define SMEM_HALFWORDS (2 * 8 * 8 * 32 * 16)   // 65536 -> 128 KB

extern "C" __global__ __launch_bounds__(256)
void liquid_cell_wmma(const float* __restrict__ x_t,
                      const float* __restrict__ state,
                      const float* __restrict__ w_in,
                      const float* __restrict__ b_in,
                      const float* __restrict__ w_rec,
                      const float* __restrict__ w_gate,
                      const float* __restrict__ b_gate,
                      const float* __restrict__ bias,
                      const float* __restrict__ log_step,
                      const float* __restrict__ ln_gamma,
                      const float* __restrict__ ln_beta,
                      float* __restrict__ out)
{
    extern __shared__ __bf16 smem[];
    const int tid = threadIdx.x;

    // ---------------- Stage weights as bf16 WMMA B-fragments in LDS ----------------
    // B-matrix (KxN=32x16 per fragment) layout: lane = n + 16*kk[3];
    // element index within lane's 16 halfwords: e = (kk & 7) + 8*kk[4]  (K-pairs packed per dword).
    for (int i = tid; i < KTOT * D_H; i += 256) {
        const int k = i & (KTOT - 1);
        const int n = i >> 8;
        const float wv = (k < D_IN) ? w_in[n * D_IN + k] : w_rec[n * D_H + (k - D_IN)];
        const int c  = k >> 5;
        const int kk = k & 31;
        const int t  = n >> 4;
        const int ln = (n & 15) + (((kk >> 3) & 1) << 4);
        smem[WD_OFF + ((((c << 3) + t) * 32 + ln) << 4) + ((kk & 7) + ((kk >> 4) << 3))] = (__bf16)wv;
    }
    for (int i = tid; i < KTOT * D_H; i += 256) {
        const int k = i & (KTOT - 1);
        const int n = i >> 8;
        const float wv = w_gate[n * KTOT + k];
        const int c  = k >> 5;
        const int kk = k & 31;
        const int t  = n >> 4;
        const int ln = (n & 15) + (((kk >> 3) & 1) << 4);
        smem[WG_OFF + ((((c << 3) + t) * 32 + ln) << 4) + ((kk & 7) + ((kk >> 4) << 3))] = (__bf16)wv;
    }
    __syncthreads();

    // ---------------- Per-wave 16-row M-tile ----------------
    const int wave  = tid >> 5;
    const int lane  = tid & 31;
    const int lhalf = lane >> 4;                       // selects kk[3] for A-fragments
    const int rowbase = (blockIdx.x * 8 + wave) * 16;
    const int row     = rowbase + (lane & 15);

    const float* xr = x_t   + (size_t)row * D_IN;
    const float* sr = state + (size_t)row * D_H;

    // Pass 1: build drive A-fragments (x | tanh(state)), accumulate LN stats.
    // A 16x32 bf16 layout: lane (row, half); vgprs 0-3 hold kk = 8*half+0..7,
    // vgprs 4-7 hold kk = 16+8*half+0..7  ->  two contiguous 8-float runs.
    v16bf Ad[8];
    float sum = 0.0f, ssq = 0.0f;
#pragma unroll
    for (int c = 0; c < 8; ++c) {
        const int koff = (c << 5) + (lhalf << 3);      // gi index of run0
        const float* src = (c < 4) ? (xr + koff) : (sr + (koff - D_IN));
        const float4 q0 = *(const float4*)(src);
        const float4 q1 = *(const float4*)(src + 4);
        const float4 q2 = *(const float4*)(src + 16);
        const float4 q3 = *(const float4*)(src + 20);
        const float vv[16] = {q0.x,q0.y,q0.z,q0.w, q1.x,q1.y,q1.z,q1.w,
                              q2.x,q2.y,q2.z,q2.w, q3.x,q3.y,q3.z,q3.w};
#pragma unroll
        for (int e = 0; e < 16; ++e) {
            const float f = vv[e];
            sum += f;
            ssq += f * f;
            Ad[c][e] = (c < 4) ? (__bf16)f : (__bf16)FAST_TANH(f);
        }
    }
    // Lanes l and l^16 hold disjoint halves of the same row's 256 gi values.
    sum += __shfl_xor(sum, 16, 32);
    ssq += __shfl_xor(ssq, 16, 32);
    const float mu = sum * (1.0f / KTOT);
    const float rs = rsqrtf(ssq * (1.0f / KTOT) - mu * mu + LN_EPSF);

    // Pass 2: build LayerNorm'ed gate A-fragments (reload from hot cache).
    v16bf Ag[8];
#pragma unroll
    for (int c = 0; c < 8; ++c) {
        const int koff = (c << 5) + (lhalf << 3);
        const float* src = (c < 4) ? (xr + koff) : (sr + (koff - D_IN));
        const float4 q0 = *(const float4*)(src);
        const float4 q1 = *(const float4*)(src + 4);
        const float4 q2 = *(const float4*)(src + 16);
        const float4 q3 = *(const float4*)(src + 20);
        const float4 g0 = *(const float4*)(ln_gamma + koff);
        const float4 g1 = *(const float4*)(ln_gamma + koff + 4);
        const float4 g2 = *(const float4*)(ln_gamma + koff + 16);
        const float4 g3 = *(const float4*)(ln_gamma + koff + 20);
        const float4 t0 = *(const float4*)(ln_beta + koff);
        const float4 t1 = *(const float4*)(ln_beta + koff + 4);
        const float4 t2 = *(const float4*)(ln_beta + koff + 16);
        const float4 t3 = *(const float4*)(ln_beta + koff + 20);
        const float vv[16] = {q0.x,q0.y,q0.z,q0.w, q1.x,q1.y,q1.z,q1.w,
                              q2.x,q2.y,q2.z,q2.w, q3.x,q3.y,q3.z,q3.w};
        const float gg[16] = {g0.x,g0.y,g0.z,g0.w, g1.x,g1.y,g1.z,g1.w,
                              g2.x,g2.y,g2.z,g2.w, g3.x,g3.y,g3.z,g3.w};
        const float bb[16] = {t0.x,t0.y,t0.z,t0.w, t1.x,t1.y,t1.z,t1.w,
                              t2.x,t2.y,t2.z,t2.w, t3.x,t3.y,t3.z,t3.w};
#pragma unroll
        for (int e = 0; e < 16; ++e) {
            Ag[c][e] = (__bf16)((vv[e] - mu) * rs * gg[e] + bb[e]);
        }
    }

    // ---------------- GEMM: 8 N-tiles x 8 K-chunks x 2 matmuls ----------------
    for (int t = 0; t < 8; ++t) {
        v8f accd = {0.f,0.f,0.f,0.f,0.f,0.f,0.f,0.f};
        v8f accg = {0.f,0.f,0.f,0.f,0.f,0.f,0.f,0.f};
#pragma unroll
        for (int c = 0; c < 8; ++c) {
            const int fro = ((((c << 3) + t) * 32 + lane) << 4);
            const v16bf bd = *(const v16bf*)(smem + WD_OFF + fro);
            const v16bf bg = *(const v16bf*)(smem + WG_OFF + fro);
            accd = __builtin_amdgcn_wmma_f32_16x16x32_bf16(false, Ad[c], false, bd,
                                                           (short)0, accd, false, false);
            accg = __builtin_amdgcn_wmma_f32_16x16x32_bf16(false, Ag[c], false, bg,
                                                           (short)0, accg, false, false);
        }
        // Epilogue. C/D layout: lanes 0-15: N=lane, M=vgpr; lanes 16-31: N=lane-16, M=vgpr+8.
        const int n = (t << 4) + (lane & 15);
        const float bsum  = b_in[n] + bias[n];
        const float bg0   = b_gate[n];
        const float stepn = FAST_SIGMOID(log_step[n]);
#pragma unroll
        for (int v = 0; v < 8; ++v) {
            const int r = rowbase + v + (lhalf << 3);
            const float target = FAST_TANH(accd[v] + bsum);
            const float gate   = FAST_SIGMOID(accg[v] + bg0);
            const float blend  = fminf(stepn * gate, 1.0f);   // DT == 1.0
            const float sold   = state[(size_t)r * D_H + n];
            // out is written once and never re-read: non-temporal store keeps
            // x/state slabs resident in WGP$/L2 for the pass-2 reload.
            __builtin_nontemporal_store(sold + blend * (target - sold),
                                        &out[(size_t)r * D_H + n]);
        }
    }
}

extern "C" void kernel_launch(void* const* d_in, const int* in_sizes, int n_in,
                              void* d_out, int out_size, void* d_ws, size_t ws_size,
                              hipStream_t stream) {
    (void)in_sizes; (void)n_in; (void)d_ws; (void)ws_size; (void)out_size;
    const float* x_t      = (const float*)d_in[0];
    const float* state    = (const float*)d_in[1];
    const float* w_in     = (const float*)d_in[2];
    const float* b_in     = (const float*)d_in[3];
    const float* w_rec    = (const float*)d_in[4];
    const float* w_gate   = (const float*)d_in[5];
    const float* b_gate   = (const float*)d_in[6];
    const float* bias_    = (const float*)d_in[7];
    const float* log_step = (const float*)d_in[8];
    const float* ln_gamma = (const float*)d_in[9];
    const float* ln_beta  = (const float*)d_in[10];
    float* out = (float*)d_out;

    const dim3 grid(NBLK / 128);   // 8 waves/block * 16 rows/wave
    const dim3 block(256);
    const size_t smem_bytes = (size_t)SMEM_HALFWORDS * sizeof(__bf16);  // 128 KB
    liquid_cell_wmma<<<grid, block, smem_bytes, stream>>>(
        x_t, state, w_in, b_in, w_rec, w_gate, b_gate, bias_,
        log_step, ln_gamma, ln_beta, out);
}
